// GraphASD_53953379173202
// MI455X (gfx1250) — matrix operand order repaired and checked by reading, not compiled, not run on hardware
//
#include <hip/hip_runtime.h>

#define NEGF (-1e30f)

typedef __attribute__((ext_vector_type(16))) _Float16 v16h;
typedef __attribute__((ext_vector_type(8)))  float    v8f;
typedef __attribute__((ext_vector_type(4)))  unsigned int u32x4;
typedef __attribute__((ext_vector_type(8)))  int      i32x8;
typedef __attribute__((ext_vector_type(4)))  int      i32x4;

#if defined(__has_builtin)
# if __has_builtin(__builtin_amdgcn_tensor_load_to_lds) && __has_builtin(__builtin_amdgcn_s_wait_tensorcnt)
#  define HAVE_TDM 1
# endif
#endif
#ifndef HAVE_TDM
# define HAVE_TDM 0
#endif

union Frag { v16h h; unsigned int u[8]; };

// K index of the p-th dword pair of a 16x32 f16 A fragment (ISA 7.12.2)
__device__ __forceinline__ int a_pair_k(int p, int kh) {
  return ((p < 4) ? (2 * p) : (8 + 2 * p)) + kh * 8;
}

__device__ __forceinline__ void atomicFMax(float* addr, float v) {
  // monotonic float ordering trick; buffer is initialized to NEGF
  if (v >= 0.0f) atomicMax((int*)addr, __float_as_int(v));
  else           atomicMin((unsigned int*)addr, __float_as_uint(v));
}

__device__ __forceinline__ int edge_mask(int kind, int a) {
  switch (kind) {
    case 0:  return a == -2;                 // m_audio
    case 1:  return (a >= 0) & (a <= 1);     // m1
    case 2:  return (a >= -1) & (a <= 0);    // m2
    case 3:  return (a >= -1) & (a <= 1);    // m3
    default: return 0;
  }
}

#if HAVE_TDM
// Issue a TDM 2D tile load (global f16 tensor -> LDS), per CDNA5 D# layout.
// Tensor: td1 rows x td0 elements (row stride stride0 elements, 2B/elem).
// Tile:   tile1 rows x tile0 elements, packed contiguously into LDS.
__device__ __forceinline__ void tdm_load_2d(void* ldsPtr, const void* gPtr,
                                            unsigned td0, unsigned td1,
                                            unsigned tile0, unsigned tile1,
                                            unsigned long long stride0) {
  unsigned long long ga = (unsigned long long)(uintptr_t)gPtr;
  unsigned lds = (unsigned)(uintptr_t)ldsPtr;  // flat->LDS: low 32 bits
  u32x4 g0 = { 1u,                                   // count=1 (valid), user mode
               lds,                                  // lds_addr
               (unsigned)ga,                         // global_addr[31:0]
               (unsigned)((ga >> 32) & 0x1ffffffull) // global_addr[56:32]
                   | 0x80000000u };                  // type=2 ("image")
  i32x8 g1 = {
    (int)0x00010000u,                                       // data_size=1 (2B)
    (int)((td0 & 0xffffu) << 16),                           // tensor_dim0[15:0]
    (int)(((td0 >> 16) & 0xffffu) | ((td1 & 0xffffu) << 16)),
    (int)(((td1 >> 16) & 0xffffu) | ((tile0 & 0xffffu) << 16)),
    (int)(tile1 & 0xffffu),                                 // tile_dim1, tile_dim2=0
    (int)(unsigned)(stride0 & 0xffffffffull),               // tensor_dim0_stride lo
    (int)(unsigned)((stride0 >> 32) & 0xffffull),           // stride hi, dim1_stride=0
    0 };
  i32x4 z4 = {0, 0, 0, 0};
#if __clang_major__ >= 23
  i32x8 z8 = {0, 0, 0, 0, 0, 0, 0, 0};
  __builtin_amdgcn_tensor_load_to_lds(g0, g1, z4, z4, z8, 0);
#else
  __builtin_amdgcn_tensor_load_to_lds(g0, g1, z4, z4, 0);
#endif
}
#endif

// ---------------------------------------------------------------- utilities
__global__ void k_fill(float* __restrict__ p, float v, int n) {
  int i = blockIdx.x * blockDim.x + threadIdx.x;
  if (i < n) p[i] = v;
}

// transpose-convert weights: W [Krows,128] f32 row-major -> out[n][k] f16
// (K-major per output column; dstK = row length of out, koff = k offset)
__global__ void k_cvt_tr(const float* __restrict__ W, _Float16* __restrict__ out,
                         int Krows, int dstK, int koff) {
  int i = blockIdx.x * blockDim.x + threadIdx.x;
  if (i >= Krows * 128) return;
  int k = i >> 7, n = i & 127;
  out[(size_t)n * dstK + koff + k] = (_Float16)W[(size_t)k * 128 + n];
}

// W1 is [256,128]; wdT = (top - bottom)^T (i-role), wqT = bottom^T (j-role)
__global__ void k_make_wd(const float* __restrict__ W1,
                          _Float16* __restrict__ wdT, _Float16* __restrict__ wqT) {
  int i = blockIdx.x * blockDim.x + threadIdx.x;  // 0..16383
  int k = i >> 7, n = i & 127;
  float top = W1[(size_t)k * 128 + n];
  float bot = W1[(size_t)(k + 128) * 128 + n];
  wdT[(size_t)n * 128 + k] = (_Float16)(top - bot);
  wqT[(size_t)n * 128 + k] = (_Float16)bot;
}

// relu(seg_max) with empty->0 over a NEG-initialized buffer == max(v,0)
__global__ void k_finalize(float* __restrict__ v, _Float16* __restrict__ out16, int n) {
  int i = blockIdx.x * blockDim.x + threadIdx.x;
  if (i >= n) return;
  float x = fmaxf(v[i], 0.0f);
  v[i] = x;
  if (out16) out16[i] = (_Float16)x;
}

// ------------------------------------------------------------- WMMA GEMM
// out[M,128] = A[M,K] @ B^T (+bias) (opt relu) (opt accumulate)
// A: f32 or f16 (flag); Bt: f16 [128][K] (K-major per output column).
// M multiple of 128, K multiple of 32.
__global__ __launch_bounds__(256)
void k_gemm(const void* __restrict__ Aptr, int a_is_f32,
            const _Float16* __restrict__ Bt,
            const float* __restrict__ bias,
            float* __restrict__ out32, _Float16* __restrict__ out16,
            int M, int K, int do_relu, int do_acc) {
  __shared__ _Float16 sA[128 * 32];   // [row][k]
  __shared__ _Float16 sB[128 * 32];   // [n][k]
  const int tid = threadIdx.x, wid = tid >> 5, lane = tid & 31;
  const int row0 = blockIdx.x * 128;
  const float*    A32 = (const float*)Aptr;
  const _Float16* A16 = (const _Float16*)Aptr;

  v8f acc[8];
  const v8f vzero = {0.f, 0.f, 0.f, 0.f, 0.f, 0.f, 0.f, 0.f};
#pragma unroll
  for (int i = 0; i < 8; ++i) acc[i] = vzero;

  const int r = tid >> 1, halfsel = tid & 1;
  const int c0 = halfsel * 16;

  for (int kk = 0; kk < K; kk += 32) {
    // stage B tile: one TDM 2D descriptor (32 elems x 128 rows, stride K)
#if HAVE_TDM
    if (wid == 0)
      tdm_load_2d(sB, Bt + kk, (unsigned)K, 128u, 32u, 128u,
                  (unsigned long long)(unsigned)K);
#else
    {
      const _Float16* s = Bt + (size_t)r * K + kk + c0;
#pragma unroll
      for (int j = 0; j < 16; ++j) sB[r * 32 + c0 + j] = s[j];
    }
#endif
    // stage A tile (convert f32->f16 on the fly)
    if (a_is_f32) {
      const float* s = A32 + (size_t)(row0 + r) * K + kk + c0;
#pragma unroll
      for (int j = 0; j < 16; ++j) sA[r * 32 + c0 + j] = (_Float16)s[j];
    } else {
      const _Float16* s = A16 + (size_t)(row0 + r) * K + kk + c0;
#pragma unroll
      for (int j = 0; j < 16; ++j) sA[r * 32 + c0 + j] = s[j];
    }
    // prefetch next A chunk (global_prefetch_b8)
    if (kk + 32 < K) {
      const char* pf = a_is_f32
          ? (const char*)(A32 + (size_t)(row0 + r) * K + kk + 32 + c0)
          : (const char*)(A16 + (size_t)(row0 + r) * K + kk + 32 + c0);
      __builtin_prefetch(pf, 0, 1);
    }
#if HAVE_TDM
    if (wid == 0) __builtin_amdgcn_s_wait_tensorcnt(0);
#endif
    __syncthreads();

    const int m = lane & 15, kh = lane >> 4;
    Frag af;
#pragma unroll
    for (int p = 0; p < 8; ++p)
      af.u[p] = *(const unsigned int*)&sA[(wid * 16 + m) * 32 + a_pair_k(p, kh)];
#pragma unroll
    for (int nt = 0; nt < 8; ++nt) {
      Frag bf;
      const int n = nt * 16 + m;
#pragma unroll
      for (int p = 0; p < 8; ++p)
        bf.u[p] = *(const unsigned int*)&sB[n * 32 + a_pair_k(p, kh)];
      acc[nt] = __builtin_amdgcn_wmma_f32_16x16x32_f16(
          false, af.h, false, bf.h, (short)0, acc[nt], false, false);
    }
    __syncthreads();
  }

  // write out: C/D layout -> lanes 0-15: M=row, lanes 16-31: M=row+8
  const int n = lane & 15, kh = lane >> 4;
#pragma unroll
  for (int nt = 0; nt < 8; ++nt) {
#pragma unroll
    for (int rr = 0; rr < 8; ++rr) {
      const int mrow = wid * 16 + kh * 8 + rr;
      const int col = nt * 16 + n;
      float v = acc[nt][rr];
      if (bias) v += bias[col];
      if (do_relu) v = fmaxf(v, 0.0f);
      const size_t o = (size_t)(row0 + mrow) * 128 + col;
      if (do_acc) v += out32[o];
      out32[o] = v;
      if (out16) out16[o] = (_Float16)v;
    }
  }
}

// ------------------------------------------------- edge scatter kernels
__global__ void k_edge_vpa(const int* __restrict__ src, const int* __restrict__ dst,
                           const int* __restrict__ attr, const float* __restrict__ x,
                           float* __restrict__ out, int E) {
  long long gid = (long long)blockIdx.x * blockDim.x + threadIdx.x;
  int e = (int)(gid >> 7), c = (int)(gid & 127);
  if (e >= E || attr[e] != -3) return;
  int s = src[e], d = dst[e];
  atomicFMax(&out[(size_t)d * 128 + c],
             x[(size_t)d * 128 + c] + x[(size_t)s * 128 + c]);
}

__global__ void k_edge_acv(const int* __restrict__ src, const int* __restrict__ dst,
                           const int* __restrict__ attr, const float* __restrict__ x,
                           const float* __restrict__ ares, float* __restrict__ out, int E) {
  long long gid = (long long)blockIdx.x * blockDim.x + threadIdx.x;
  int e = (int)(gid >> 7), c = (int)(gid & 127);
  if (e >= E || attr[e] != 3) return;
  int s = src[e], d = dst[e];
  float v = ares[s] * x[(size_t)s * 128 + c] + x[(size_t)d * 128 + c];
  atomicAdd(&out[(size_t)d * 128 + c], v);
}

__global__ void k_edge_cnt(const int* __restrict__ attr, const int* __restrict__ dst,
                           int kind, float* __restrict__ cnt, int E) {
  int e = blockIdx.x * blockDim.x + threadIdx.x;
  if (e >= E || !edge_mask(kind, attr[e])) return;
  atomicAdd(&cnt[dst[e]], 1.0f);
}

__global__ void k_edge_sum(const int* __restrict__ src, const int* __restrict__ dst,
                           const int* __restrict__ attr, int kind,
                           const float* __restrict__ h, float* __restrict__ s, int E) {
  long long gid = (long long)blockIdx.x * blockDim.x + threadIdx.x;
  int e = (int)(gid >> 7), c = (int)(gid & 127);
  if (e >= E || !edge_mask(kind, attr[e])) return;
  atomicAdd(&s[(size_t)dst[e] * 128 + c], h[(size_t)src[e] * 128 + c]);
}

// --------------------------------------- EdgeConv layer-2 (per-edge WMMA)
// h_e = relu(P[dst] + Q[src]); D = h_e @ W2 + b2; scatter-max into out[dst].
// 16-edge tiles; W2^T ([n][k] f16) staged in LDS once per block via TDM.
__global__ __launch_bounds__(128)
void k_edge_mlp(const int* __restrict__ src, const int* __restrict__ dst,
                const int* __restrict__ attr, int kind,
                const float* __restrict__ P, const float* __restrict__ Q,
                const _Float16* __restrict__ W2t, const float* __restrict__ b2,
                float* __restrict__ out, int E) {
  __shared__ _Float16 sB[128 * 128];  // W2^T [n][k]
  __shared__ _Float16 sA[16 * 128];
  __shared__ float    sD[16 * 128];
  __shared__ int sSrc[16], sDst[16], sVal[16];
  const int tid = threadIdx.x, wid = tid >> 5, lane = tid & 31;

  // stage the whole 128x128 W2^T once per block
#if HAVE_TDM
  if (wid == 0) {
    tdm_load_2d(sB, W2t, 128u, 128u, 128u, 128u, 128ull);
    __builtin_amdgcn_s_wait_tensorcnt(0);
  }
#else
  {
    const int n = tid;
#pragma unroll 4
    for (int k = 0; k < 128; ++k) sB[n * 128 + k] = W2t[(size_t)n * 128 + k];
  }
#endif
  __syncthreads();

  const int ntiles = E >> 4;
  for (int t = blockIdx.x; t < ntiles; t += gridDim.x) {
    if (tid < 16) {
      int e = t * 16 + tid;
      sSrc[tid] = src[e];
      sDst[tid] = dst[e];
      sVal[tid] = edge_mask(kind, attr[e]);
    }
    __syncthreads();
    // build A tile: 16 edges x 128 f16, zero rows for masked-out edges
    {
      const int rr = tid >> 3;
      const int c0 = (tid & 7) * 16;
      const int s = sSrc[rr], d = sDst[rr], ok = sVal[rr];
#pragma unroll
      for (int j = 0; j < 16; ++j) {
        float h = 0.0f;
        if (ok)
          h = fmaxf(P[(size_t)d * 128 + c0 + j] + Q[(size_t)s * 128 + c0 + j], 0.0f);
        sA[rr * 128 + c0 + j] = (_Float16)h;
      }
    }
    __syncthreads();
    // each of 4 waves computes 2 N-tiles (32 output cols)
    {
      const int m = lane & 15, kh = lane >> 4;
      const v8f vzero = {0.f, 0.f, 0.f, 0.f, 0.f, 0.f, 0.f, 0.f};
      v8f acc0 = vzero, acc1 = vzero;
#pragma unroll
      for (int kc = 0; kc < 4; ++kc) {
        const int kb = kc * 32;
        Frag af;
#pragma unroll
        for (int p = 0; p < 8; ++p)
          af.u[p] = *(const unsigned int*)&sA[m * 128 + kb + a_pair_k(p, kh)];
        Frag bf0, bf1;
        const int n0 = (2 * wid) * 16 + m, n1 = (2 * wid + 1) * 16 + m;
#pragma unroll
        for (int p = 0; p < 8; ++p) {
          bf0.u[p] = *(const unsigned int*)&sB[n0 * 128 + kb + a_pair_k(p, kh)];
          bf1.u[p] = *(const unsigned int*)&sB[n1 * 128 + kb + a_pair_k(p, kh)];
        }
        acc0 = __builtin_amdgcn_wmma_f32_16x16x32_f16(false, af.h, false, bf0.h,
                                                      (short)0, acc0, false, false);
        acc1 = __builtin_amdgcn_wmma_f32_16x16x32_f16(false, af.h, false, bf1.h,
                                                      (short)0, acc1, false, false);
      }
      const int c0 = (2 * wid) * 16 + m, c1 = (2 * wid + 1) * 16 + m;
#pragma unroll
      for (int rr = 0; rr < 8; ++rr) {
        const int mrow = kh * 8 + rr;
        sD[mrow * 128 + c0] = acc0[rr] + b2[c0];
        sD[mrow * 128 + c1] = acc1[rr] + b2[c1];
      }
    }
    __syncthreads();
    // scatter-max rows to destination nodes
    {
      const int rr = tid >> 3;
      const int c0 = (tid & 7) * 16;
      if (sVal[rr]) {
        const int d = sDst[rr];
#pragma unroll
        for (int j = 0; j < 16; ++j)
          atomicFMax(&out[(size_t)d * 128 + c0 + j], sD[rr * 128 + c0 + j]);
      }
    }
    __syncthreads();
  }
}

// per-node dot: ares[n] = a[n,:] . fcW + fcb  (one wave32 per node)
__global__ __launch_bounds__(256)
void k_fc(const float* __restrict__ a, const float* __restrict__ fw,
          const float* __restrict__ fb, float* __restrict__ ares, int n) {
  const int lane = threadIdx.x & 31, wid = threadIdx.x >> 5;
  const int node = blockIdx.x * 8 + wid;
  if (node >= n) return;
  float v = 0.0f;
#pragma unroll
  for (int j = lane; j < 128; j += 32) v += a[(size_t)node * 128 + j] * fw[j];
  for (int off = 16; off; off >>= 1) v += __shfl_down(v, off, 32);
  if (lane == 0) ares[node] = v + fb[0];
}

// cat[mean_j h_j, h_i] as f16 for the fused SAGE GEMM (K=256)
__global__ void k_mean_cat(const float* __restrict__ s, const float* __restrict__ cnt,
                           const _Float16* __restrict__ h16, _Float16* __restrict__ cat,
                           int n) {
  int gid = blockIdx.x * blockDim.x + threadIdx.x;
  if (gid >= n * 128) return;
  int node = gid >> 7, c = gid & 127;
  float cd = fmaxf(cnt[node], 1.0f);
  cat[(size_t)node * 256 + c] = (_Float16)(s[gid] / cd);
  cat[(size_t)node * 256 + 128 + c] = h16[gid];
}

// out[i*T + t, c] = xo[(i*spk*T + t), c]  with T=128
__global__ void k_gather(const float* __restrict__ xo, float* __restrict__ out,
                         const int* __restrict__ spkp, int total) {
  int gid = blockIdx.x * blockDim.x + threadIdx.x;
  if (gid >= total) return;
  const int spk = *spkp;
  const int c = gid & 127;
  const int t = (gid >> 7) & 127;
  const int i = gid >> 14;
  const int node = i * spk * 128 + t;
  out[gid] = xo[(size_t)node * 128 + c];
}

// ================================================================= launch
extern "C" void kernel_launch(void* const* d_in, const int* in_sizes, int n_in,
                              void* d_out, int out_size, void* d_ws, size_t ws_size,
                              hipStream_t stream) {
  const int Nn = 32768, Ee = 262144, FD = 1024;
  const int NVT = 24576, NAT = 8192;

  const float* x_visual = (const float*)d_in[0];
  const float* x_audio  = (const float*)d_in[1];
  const float* W011 = (const float*)d_in[2];  const float* b011 = (const float*)d_in[3];
  const float* W012 = (const float*)d_in[4];  const float* b012 = (const float*)d_in[5];
  const float* ecW1[4] = {(const float*)d_in[6],  (const float*)d_in[10],
                          (const float*)d_in[14], (const float*)d_in[18]};
  const float* ecB1[4] = {(const float*)d_in[7],  (const float*)d_in[11],
                          (const float*)d_in[15], (const float*)d_in[19]};
  const float* ecW2[4] = {(const float*)d_in[8],  (const float*)d_in[12],
                          (const float*)d_in[16], (const float*)d_in[20]};
  const float* ecB2[4] = {(const float*)d_in[9],  (const float*)d_in[13],
                          (const float*)d_in[17], (const float*)d_in[21]};
  const float* sage_Wl = (const float*)d_in[22];
  const float* sage_bl = (const float*)d_in[23];
  const float* sage_Wr = (const float*)d_in[24];
  const float* fc_W = (const float*)d_in[25];
  const float* fc_b = (const float*)d_in[26];
  const int* ei   = (const int*)d_in[27];
  const int* attr = (const int*)d_in[28];
  const int* spk  = (const int*)d_in[29];
  const int* esrc = ei;
  const int* edst = ei + Ee;

  size_t off = 0;
  auto alloc = [&](size_t b) -> void* {
    void* p = (char*)d_ws + off;
    off += (b + 255) & ~(size_t)255;
    return p;
  };
  float*     xf32    = (float*)alloc((size_t)Nn * 128 * 4);
  float*     xag     = (float*)alloc((size_t)Nn * 128 * 4);
  _Float16*  xag16   = (_Float16*)alloc((size_t)Nn * 128 * 2);
  float*     P       = (float*)alloc((size_t)Nn * 128 * 4);
  float*     Q       = (float*)alloc((size_t)Nn * 128 * 4);
  float*     ecout   = (float*)alloc((size_t)Nn * 128 * 4);
  _Float16*  ecout16 = (_Float16*)alloc((size_t)Nn * 128 * 2);
  float*     ares    = (float*)alloc((size_t)Nn * 4);
  float*     xv2     = (float*)alloc((size_t)Nn * 128 * 4);
  _Float16*  xv2_16  = (_Float16*)alloc((size_t)Nn * 128 * 2);
  float*     ssum    = (float*)alloc((size_t)Nn * 128 * 4);
  float*     cnt     = (float*)alloc((size_t)Nn * 4);
  _Float16*  cat16   = (_Float16*)alloc((size_t)Nn * 256 * 2);
  float*     xo      = (float*)alloc((size_t)Nn * 128 * 4);
  _Float16*  wproj   = (_Float16*)alloc((size_t)FD * 128 * 2);   // [n][1024]
  _Float16*  wd      = (_Float16*)alloc((size_t)128 * 128 * 2);  // [n][128]
  _Float16*  wq      = (_Float16*)alloc((size_t)128 * 128 * 2);  // [n][128]
  _Float16*  w2      = (_Float16*)alloc((size_t)128 * 128 * 2);  // [n][128]
  _Float16*  wcat    = (_Float16*)alloc((size_t)256 * 128 * 2);  // [n][256]

  const int NC = Nn * 128;       // 4194304
  const int gNC = NC / 256;
  const int gEC = (Ee * 128) / 256;

  // 1) input projections -> x [N,128]
  k_cvt_tr<<<(FD * 128) / 256, 256, 0, stream>>>(W011, wproj, FD, FD, 0);
  k_gemm<<<NVT / 128, 256, 0, stream>>>(x_visual, 1, wproj, b011,
                                        xf32, (_Float16*)nullptr, NVT, FD, 0, 0);
  k_cvt_tr<<<(FD * 128) / 256, 256, 0, stream>>>(W012, wproj, FD, FD, 0);
  k_gemm<<<NAT / 128, 256, 0, stream>>>(x_audio, 1, wproj, b012,
                                        xf32 + (size_t)NVT * 128, (_Float16*)nullptr,
                                        NAT, FD, 0, 0);

  // 2) xa_g = relu(seg_max over vpa edges of x_d + x_s)
  k_fill<<<gNC, 256, 0, stream>>>(xag, NEGF, NC);
  k_edge_vpa<<<gEC, 256, 0, stream>>>(esrc, edst, attr, xf32, xag, Ee);
  k_finalize<<<gNC, 256, 0, stream>>>(xag, xag16, NC);

  // 3) ecA EdgeConv (mask: attr == -2)  ->  a (in ecout)
  k_make_wd<<<64, 256, 0, stream>>>(ecW1[0], wd, wq);
  k_cvt_tr<<<64, 256, 0, stream>>>(ecW2[0], w2, 128, 128, 0);
  k_gemm<<<Nn / 128, 256, 0, stream>>>(xag16, 0, wd, ecB1[0], P, (_Float16*)nullptr, Nn, 128, 0, 0);
  k_gemm<<<Nn / 128, 256, 0, stream>>>(xag16, 0, wq, (const float*)nullptr, Q, (_Float16*)nullptr, Nn, 128, 0, 0);
  k_fill<<<gNC, 256, 0, stream>>>(ecout, NEGF, NC);
  k_edge_mlp<<<2048, 128, 0, stream>>>(esrc, edst, attr, 0, P, Q, w2, ecB2[0], ecout, Ee);
  k_finalize<<<gNC, 256, 0, stream>>>(ecout, (_Float16*)nullptr, NC);

  // 4) a_res = a @ fc_W + fc_b
  k_fc<<<Nn / 8, 256, 0, stream>>>(ecout, fc_W, fc_b, ares, Nn);

  // 5) xv2 = relu(seg_sum over acv edges of a_res[s]*x_s + x_d)
  k_fill<<<gNC, 256, 0, stream>>>(xv2, 0.0f, NC);
  k_edge_acv<<<gEC, 256, 0, stream>>>(esrc, edst, attr, xf32, ares, xv2, Ee);
  k_finalize<<<gNC, 256, 0, stream>>>(xv2, xv2_16, NC);

  // SAGE fused weights [Wl; Wr]^T -> one K=256 GEMM
  k_cvt_tr<<<64, 256, 0, stream>>>(sage_Wl, wcat, 128, 256, 0);
  k_cvt_tr<<<64, 256, 0, stream>>>(sage_Wr, wcat, 128, 256, 128);
  k_fill<<<gNC, 256, 0, stream>>>(xo, 0.0f, NC);

  // 6) three branches: EdgeConv -> SAGEConv, accumulated into xo
  for (int b = 1; b <= 3; ++b) {
    k_make_wd<<<64, 256, 0, stream>>>(ecW1[b], wd, wq);
    k_cvt_tr<<<64, 256, 0, stream>>>(ecW2[b], w2, 128, 128, 0);
    k_gemm<<<Nn / 128, 256, 0, stream>>>(xv2_16, 0, wd, ecB1[b], P, (_Float16*)nullptr, Nn, 128, 0, 0);
    k_gemm<<<Nn / 128, 256, 0, stream>>>(xv2_16, 0, wq, (const float*)nullptr, Q, (_Float16*)nullptr, Nn, 128, 0, 0);
    k_fill<<<gNC, 256, 0, stream>>>(ecout, NEGF, NC);
    k_edge_mlp<<<2048, 128, 0, stream>>>(esrc, edst, attr, b, P, Q, w2, ecB2[b], ecout, Ee);
    k_finalize<<<gNC, 256, 0, stream>>>(ecout, ecout16, NC);

    k_fill<<<gNC, 256, 0, stream>>>(ssum, 0.0f, NC);
    k_fill<<<Nn / 256, 256, 0, stream>>>(cnt, 0.0f, Nn);
    k_edge_cnt<<<Ee / 256, 256, 0, stream>>>(attr, edst, b, cnt, Ee);
    k_edge_sum<<<gEC, 256, 0, stream>>>(esrc, edst, attr, b, ecout, ssum, Ee);
    k_mean_cat<<<gNC, 256, 0, stream>>>(ssum, cnt, ecout16, cat16, Nn);
    // xo += relu(cat @ [Wl;Wr] + bl)
    k_gemm<<<Nn / 128, 256, 0, stream>>>(cat16, 0, wcat, sage_bl, xo, (_Float16*)nullptr, Nn, 256, 1, 1);
  }

  // 7) gather main-speaker visual nodes
  k_gather<<<out_size / 256, 256, 0, stream>>>(xo, (float*)d_out, spk, out_size);
}
